// MultiHeadAttentionBlock_8005819040400
// MI455X (gfx1250) — matrix operand order repaired and checked by reading, not compiled, not run on hardware
//
#include <hip/hip_runtime.h>
#include <hip/hip_bf16.h>

// ---------------- problem constants ----------------
static constexpr int BB   = 4;      // batch
static constexpr int SS   = 2048;   // sequence
static constexpr int DM   = 1024;   // d_model
static constexpr int HH   = 16;     // heads
static constexpr int DK   = 64;     // head dim

typedef __attribute__((ext_vector_type(16))) __bf16 v16bf;
typedef __attribute__((ext_vector_type(8)))  __bf16 v8bf;
typedef __attribute__((ext_vector_type(8)))  float  v8f;

static __device__ __forceinline__ __bf16 f2bf(float f) {
    union { float f; unsigned u; } a; a.f = f;
    unsigned u = a.u + 0x7FFFu + ((a.u >> 16) & 1u);   // round-to-nearest-even
    union { unsigned short s; __bf16 b; } r; r.s = (unsigned short)(u >> 16);
    return r.b;
}

// ======================================================================
// f32 -> bf16 bulk convert (8 elements / thread, fully coalesced b128)
// ======================================================================
__global__ void __launch_bounds__(256)
cvt_bf16_kernel(const float* __restrict__ in, __bf16* __restrict__ out) {
    const size_t i = ((size_t)blockIdx.x * blockDim.x + threadIdx.x) * 8;
    const float4 a = *(const float4*)(in + i);
    const float4 b = *(const float4*)(in + i + 4);
    v8bf o;
    o[0] = f2bf(a.x); o[1] = f2bf(a.y); o[2] = f2bf(a.z); o[3] = f2bf(a.w);
    o[4] = f2bf(b.x); o[5] = f2bf(b.y); o[6] = f2bf(b.z); o[7] = f2bf(b.w);
    *(v8bf*)(out + i) = o;
}

// ======================================================================
// Tiled bf16 WMMA GEMM:  out[m][n] = sum_k A[m][k] * W[n][k] + bias[n]
// A: (M x 1024) bf16 row-major, W: (1024 x 1024) bf16 row-major (out,in).
// MODE 0: bf16 out, head-major   ((b*H+h)*S+s)*DK+d
// MODE 1: f32  out, row-major    m*DM+n
// MODE 2: bf16 out, V-transposed ((b*H+h)*DK+d)*S+s
// 256 thr (8 waves, 4Mx2N), WG tile 128x128, wave tile 32x64, K-step 32.
// As is stored plain (k-contiguous) so each lane's A-fragment is one contiguous
// 32B read (k = 16*h + s, the same order as the B-fragment).
// ======================================================================
template <int MODE>
__global__ void __launch_bounds__(256)
gemm_bias_kernel(const __bf16* __restrict__ A, const __bf16* __restrict__ W,
                 const float* __restrict__ bias, void* __restrict__ outv) {
    constexpr int TM = 128, TN = 128, TK = 32;
    __shared__ __align__(32) __bf16 As[TM][TK];   // plain
    __shared__ __align__(32) __bf16 Ws[TN][TK];   // plain

    const int tid  = threadIdx.x;
    const int wave = tid >> 5, lane = tid & 31;
    const int lm = lane & 15, lk = lane >> 4;
    const int wm = wave >> 1, wn = wave & 1;     // 4(M) x 2(N) wave grid
    const int mbase = blockIdx.y * TM;
    const int nbase = blockIdx.x * TN;

    // staging: each thread copies 16 bf16 (one 32B global read) per matrix
    const int sr   = (tid * 16) >> 5;            // row 0..127
    const int scol = (tid * 16) & 31;            // 0 or 16
    const int sd0  = scol >> 1;                  // unused: 0 or 8

    v8f acc[2][4] = {};

    for (int k0 = 0; k0 < DM; k0 += TK) {
        {
            const __bf16* ap = A + (size_t)(mbase + sr) * DM + k0 + scol;
            v8bf a0 = *(const v8bf*)ap;
            v8bf a1 = *(const v8bf*)(ap + 8);
            *(v8bf*)&As[sr][scol]     = a0;      // c=scol..scol+7
            *(v8bf*)&As[sr][scol + 8] = a1;      // c=scol+8..scol+15
            const __bf16* wp = W + (size_t)(nbase + sr) * DM + k0 + scol;
            *(v8bf*)&Ws[sr][scol]     = *(const v8bf*)wp;
            *(v8bf*)&Ws[sr][scol + 8] = *(const v8bf*)(wp + 8);
        }
        __syncthreads();

        v16bf afr[2], bfr[4];
        #pragma unroll
        for (int br = 0; br < 2; ++br)           // contiguous A frag
            afr[br] = *(const v16bf*)&As[wm * 32 + br * 16 + lm][lk * 16];
        #pragma unroll
        for (int bc = 0; bc < 4; ++bc)           // contiguous B frag
            bfr[bc] = *(const v16bf*)&Ws[wn * 64 + bc * 16 + lm][lk * 16];

        #pragma unroll
        for (int br = 0; br < 2; ++br)
            #pragma unroll
            for (int bc = 0; bc < 4; ++bc)
                acc[br][bc] = __builtin_amdgcn_wmma_f32_16x16x32_bf16(
                    false, afr[br], false, bfr[bc], (short)0, acc[br][bc], false, false);
        __syncthreads();
    }

    // epilogue: C layout -> lane n = lm, row m = r + 8*lk
    #pragma unroll
    for (int bc = 0; bc < 4; ++bc) {
        const int nglob = nbase + wn * 64 + bc * 16 + lm;
        const float bv = bias[nglob];
        #pragma unroll
        for (int br = 0; br < 2; ++br) {
            #pragma unroll
            for (int r = 0; r < 8; ++r) {
                const int mglob = mbase + wm * 32 + br * 16 + r + 8 * lk;
                const float val = acc[br][bc][r] + bv;
                if (MODE == 1) {
                    ((float*)outv)[(size_t)mglob * DM + nglob] = val;
                } else {
                    const int b = mglob >> 11, s = mglob & (SS - 1);
                    const int h = nglob >> 6,  d = nglob & (DK - 1);
                    __bf16* o = (__bf16*)outv;
                    if (MODE == 0)
                        o[((size_t)(b * HH + h) * SS + s) * DK + d] = f2bf(val);
                    else
                        o[((size_t)(b * HH + h) * DK + d) * SS + s] = f2bf(val);
                }
            }
        }
    }
}

// ======================================================================
// Flash attention: each wave owns 16 query rows of one (b,h); streams keys
// 64 at a time (4 subtiles); base-2 online softmax (one butterfly + one
// rescale per 64 keys); P kept in a key-contiguous LDS tile so the P
// A-fragments are contiguous 32B reads; P*V via WMMA. ctx written bf16.
// block = 128 threads (4 waves), grid = (S/64, H, B).
// ======================================================================
__global__ void __launch_bounds__(128)
attn_kernel(const __bf16* __restrict__ Qh, const __bf16* __restrict__ Kh,
            const __bf16* __restrict__ Vt, __bf16* __restrict__ ctx) {
    __shared__ __align__(32) __bf16 Pbuf[4][16][64];   // per-wave, plain key cols

    const int wave = threadIdx.x >> 5, lane = threadIdx.x & 31;
    const int lm = lane & 15, lk = lane >> 4;
    const int qtile = blockIdx.x * 4 + wave;           // 0..127 (16 rows each)
    const int h = blockIdx.y, b = blockIdx.z;

    const __bf16* Qp = Qh + ((size_t)(b * HH + h) * SS + qtile * 16) * DK;
    const __bf16* Kp = Kh + (size_t)(b * HH + h) * SS * DK;
    const __bf16* Vp = Vt + (size_t)(b * HH + h) * DK * SS;

    // Q A-fragments for contraction dims 0..31 and 32..63 (kept in regs)
    v16bf aQ0, aQ1;
    {
        const __bf16* qrow = Qp + (size_t)lm * DK;
        v8bf l0 = *(const v8bf*)(qrow + lk * 16);
        v8bf h0 = *(const v8bf*)(qrow + lk * 16 + 8);
        aQ0 = __builtin_shufflevector(l0, h0, 0,1,2,3,4,5,6,7,8,9,10,11,12,13,14,15);
        v8bf l1 = *(const v8bf*)(qrow + 32 + lk * 16);
        v8bf h1 = *(const v8bf*)(qrow + 32 + lk * 16 + 8);
        aQ1 = __builtin_shufflevector(l1, h1, 0,1,2,3,4,5,6,7,8,9,10,11,12,13,14,15);
    }

    float mrun[8], lrun[8];
    #pragma unroll
    for (int r = 0; r < 8; ++r) { mrun[r] = -1e30f; lrun[r] = 0.0f; }
    v8f oacc[4] = {};

    // base-2 domain: scores scaled by (1/sqrt(dk)) * log2(e)
    constexpr float SC2 = 0.125f * 1.44269504088896340736f;

    // plain (contiguous) column positions for the P stores (row m = r + 8*lk, key = lm)
    const int p0pos = lm;                              // keys  0..15 of group
    const int p1pos = lm | 16;                         // keys 16..31 of group

    for (int kt = 0; kt < SS; kt += 64) {
        // ---- scores: four 16-key subtiles (8 WMMAs) ----
        v8f sc[4];
        #pragma unroll
        for (int sub = 0; sub < 4; ++sub) {
            const __bf16* krow = Kp + (size_t)(kt + sub * 16 + lm) * DK;
            v16bf b0 = *(const v16bf*)(krow + lk * 16);        // dims 0..31
            v16bf b1 = *(const v16bf*)(krow + 32 + lk * 16);   // dims 32..63
            v8f c = {};
            c = __builtin_amdgcn_wmma_f32_16x16x32_bf16(false, aQ0, false, b0, (short)0, c, false, false);
            c = __builtin_amdgcn_wmma_f32_16x16x32_bf16(false, aQ1, false, b1, (short)0, c, false, false);
            sc[sub] = c;
        }

        // ---- online softmax (row m = r + 8*lk) ----
        #pragma unroll
        for (int r = 0; r < 8; ++r) {
            float v0 = sc[0][r] * SC2, v1 = sc[1][r] * SC2;
            float v2 = sc[2][r] * SC2, v3 = sc[3][r] * SC2;
            float rowm = fmaxf(fmaxf(v0, v1), fmaxf(v2, v3));
            #pragma unroll
            for (int msk = 1; msk < 16; msk <<= 1)
                rowm = fmaxf(rowm, __shfl_xor(rowm, msk, 32));
            const float mnew = fmaxf(mrun[r], rowm);
            const float corr = exp2f(mrun[r] - mnew);
            mrun[r] = mnew;
            const float p0 = exp2f(v0 - mnew), p1 = exp2f(v1 - mnew);
            const float p2 = exp2f(v2 - mnew), p3 = exp2f(v3 - mnew);
            lrun[r] = lrun[r] * corr + ((p0 + p1) + (p2 + p3));
            #pragma unroll
            for (int nb = 0; nb < 4; ++nb) oacc[nb][r] *= corr;
            const int pr = r + 8 * lk;
            Pbuf[wave][pr][p0pos]      = f2bf(p0);   // group 0 (keys 0..31)
            Pbuf[wave][pr][p1pos]      = f2bf(p1);
            Pbuf[wave][pr][32 + p0pos] = f2bf(p2);   // group 1 (keys 32..63)
            Pbuf[wave][pr][32 + p1pos] = f2bf(p3);
        }

        // ---- P (16x64) x V (64x64): 8 WMMAs, contiguous frag loads ----
        v16bf aP0 = *(const v16bf*)&Pbuf[wave][lm][lk * 16];
        v16bf aP1 = *(const v16bf*)&Pbuf[wave][lm][32 + lk * 16];
        #pragma unroll
        for (int nb = 0; nb < 4; ++nb) {
            const __bf16* vrow = Vp + (size_t)(nb * 16 + lm) * SS + kt;  // d-major
            v16bf bv0 = *(const v16bf*)(vrow + lk * 16);
            v16bf bv1 = *(const v16bf*)(vrow + 32 + lk * 16);
            oacc[nb] = __builtin_amdgcn_wmma_f32_16x16x32_bf16(
                false, aP0, false, bv0, (short)0, oacc[nb], false, false);
            oacc[nb] = __builtin_amdgcn_wmma_f32_16x16x32_bf16(
                false, aP1, false, bv1, (short)0, oacc[nb], false, false);
        }
    }

    // ---- finalize: reduce row sums across the 16-lane half, normalize ----
    #pragma unroll
    for (int r = 0; r < 8; ++r) {
        float l = lrun[r];
        #pragma unroll
        for (int msk = 1; msk < 16; msk <<= 1) l += __shfl_xor(l, msk, 32);
        lrun[r] = 1.0f / l;
    }
    #pragma unroll
    for (int nb = 0; nb < 4; ++nb) {
        const int d = nb * 16 + lm;
        #pragma unroll
        for (int r = 0; r < 8; ++r) {
            const int srow = qtile * 16 + r + 8 * lk;
            ctx[((size_t)b * SS + srow) * DM + h * DK + d] = f2bf(oacc[nb][r] * lrun[r]);
        }
    }
}

// ======================================================================
extern "C" void kernel_launch(void* const* d_in, const int* in_sizes, int n_in,
                              void* d_out, int out_size, void* d_ws, size_t ws_size,
                              hipStream_t stream) {
    (void)in_sizes; (void)n_in; (void)out_size; (void)ws_size;
    const float* q  = (const float*)d_in[0];
    const float* k  = (const float*)d_in[1];
    const float* v  = (const float*)d_in[2];
    // d_in[3] = mask (no-op in the reference)
    const float* wq = (const float*)d_in[4];  const float* bq = (const float*)d_in[5];
    const float* wk = (const float*)d_in[6];  const float* bk = (const float*)d_in[7];
    const float* wv = (const float*)d_in[8];  const float* bv = (const float*)d_in[9];
    const float* wo = (const float*)d_in[10]; const float* bo = (const float*)d_in[11];

    char* ws = (char*)d_ws;
    const size_t ACT = (size_t)BB * SS * DM;                 // 8M elements
    const size_t WEL = (size_t)DM * DM;                      // 1M elements
    __bf16* qh   = (__bf16*)(ws);                            // 16 MB
    __bf16* kh   = (__bf16*)(ws + ACT * 2);                  // 16 MB
    __bf16* vt   = (__bf16*)(ws + ACT * 4);                  // 16 MB
    __bf16* ctxb = (__bf16*)(ws + ACT * 6);                  // 16 MB
    __bf16* abuf = (__bf16*)(ws + ACT * 8);                  // 16 MB (recycled)
    __bf16* wbuf = (__bf16*)(ws + ACT * 10);                 //  2 MB (recycled)

    const dim3 gGemm(DM / 128, (BB * SS) / 128);
    const int  gCvtA = (int)(ACT / 8 / 256);                 // activation convert
    const int  gCvtW = (int)(WEL / 8 / 256);                 // weight convert

    // Q projection
    cvt_bf16_kernel<<<gCvtA, 256, 0, stream>>>(q,  abuf);
    cvt_bf16_kernel<<<gCvtW, 256, 0, stream>>>(wq, wbuf);
    gemm_bias_kernel<0><<<gGemm, 256, 0, stream>>>(abuf, wbuf, bq, (void*)qh);
    // K projection (reuse abuf/wbuf; same-stream serialization makes this safe)
    cvt_bf16_kernel<<<gCvtA, 256, 0, stream>>>(k,  abuf);
    cvt_bf16_kernel<<<gCvtW, 256, 0, stream>>>(wk, wbuf);
    gemm_bias_kernel<0><<<gGemm, 256, 0, stream>>>(abuf, wbuf, bk, (void*)kh);
    // V projection (transposed head-major output)
    cvt_bf16_kernel<<<gCvtA, 256, 0, stream>>>(v,  abuf);
    cvt_bf16_kernel<<<gCvtW, 256, 0, stream>>>(wv, wbuf);
    gemm_bias_kernel<2><<<gGemm, 256, 0, stream>>>(abuf, wbuf, bv, (void*)vt);

    // attention
    const dim3 gAttn(SS / 64, HH, BB);
    attn_kernel<<<gAttn, 128, 0, stream>>>(qh, kh, vt, ctxb);

    // output projection (bf16 ctx -> f32 out)
    cvt_bf16_kernel<<<gCvtW, 256, 0, stream>>>(wo, wbuf);
    gemm_bias_kernel<1><<<gGemm, 256, 0, stream>>>(ctxb, wbuf, bo, d_out);
}
